// MoE_39487929319969
// MI455X (gfx1250) — matrix operand order repaired and checked by reading, not compiled, not run on hardware
//
#include <hip/hip_runtime.h>
#include <hip/hip_bf16.h>

#define B_TOK  8192
#define DIN    1024
#define DOUT   1024
#define NEXP   16
#define MAXN   (2 * B_TOK)      // worst-case assignments per expert
#define TM     128              // token rows per block
#define MAX_MT (MAXN / TM)      // 128 m-tiles per expert worst case
#define TN     128              // columns per block

typedef __attribute__((ext_vector_type(16))) __bf16 v16bf;
typedef __attribute__((ext_vector_type(8)))  float  v8f;
typedef __attribute__((ext_vector_type(2)))  __bf16 bf16x2;
typedef __attribute__((ext_vector_type(2)))  float  f32x2;

// two f32 -> packed bf16 pair -> single dword LDS store (v_cvt_pk_bf16_f32)
__device__ __forceinline__ void store_pair(__bf16* dst, float lo, float hi) {
    f32x2 f;
    f.x = lo;
    f.y = hi;
    *(bf16x2*)dst = __builtin_convertvector(f, bf16x2);
}

// ---------------------------------------------------------------------------
// Kernel 1: gating (softmax over 16 logits), top-2 selection, routing lists.
// One wave (32 lanes) per token; 8 tokens per 256-thread block.
// ---------------------------------------------------------------------------
__global__ __launch_bounds__(256) void moe_gate_route(
    const float* __restrict__ x, const float* __restrict__ gw,
    const float* __restrict__ gb,
    int* __restrict__ counts, int* __restrict__ tok_list,
    float* __restrict__ w_list)
{
    const int tid   = threadIdx.x;
    const int lane  = tid & 31;
    const int token = blockIdx.x * 8 + (tid >> 5);

    const float4* xr = (const float4*)(x + (size_t)token * DIN);

    float acc[NEXP];
#pragma unroll
    for (int e = 0; e < NEXP; ++e) acc[e] = 0.f;

#pragma unroll
    for (int c = 0; c < DIN / 128; ++c) {
        const int idx = c * 32 + lane;
        const float4 xv = xr[idx];
#pragma unroll
        for (int e = 0; e < NEXP; ++e) {
            const float4 gv = ((const float4*)(gw + (size_t)e * DIN))[idx];
            acc[e] += xv.x * gv.x + xv.y * gv.y + xv.z * gv.z + xv.w * gv.w;
        }
    }
#pragma unroll
    for (int e = 0; e < NEXP; ++e) {
#pragma unroll
        for (int off = 16; off > 0; off >>= 1)
            acc[e] += __shfl_xor(acc[e], off, 32);
    }

    if (lane == 0) {
        float logit[NEXP];
        float mx = -3.4e38f;
#pragma unroll
        for (int e = 0; e < NEXP; ++e) {
            logit[e] = acc[e] + gb[e];
            mx = fmaxf(mx, logit[e]);
        }
        float s = 0.f;
#pragma unroll
        for (int e = 0; e < NEXP; ++e) s += expf(logit[e] - mx);

        int i1 = 0; float v1 = -3.4e38f;
#pragma unroll
        for (int e = 0; e < NEXP; ++e)
            if (logit[e] > v1) { v1 = logit[e]; i1 = e; }
        int i2 = 0; float v2 = -3.4e38f;
#pragma unroll
        for (int e = 0; e < NEXP; ++e)
            if (e != i1 && logit[e] > v2) { v2 = logit[e]; i2 = e; }

        const float w1 = expf(v1 - mx) / s;
        const float w2 = expf(v2 - mx) / s;

        int p1 = atomicAdd(&counts[i1], 1);
        tok_list[i1 * MAXN + p1] = token;
        w_list [i1 * MAXN + p1] = w1;
        int p2 = atomicAdd(&counts[i2], 1);
        tok_list[i2 * MAXN + p2] = token;
        w_list [i2 * MAXN + p2] = w2;
    }
}

// ---------------------------------------------------------------------------
// Kernel 2: grouped GEMM over routed tokens, bf16 WMMA, f32 accumulate.
// Block tile: M=128 x N=128, K stepped by 32, software-pipelined.
// Wave grid 2(m) x 4(n); each wave computes a 64x32 register tile:
// 4 M-tiles x 2 N-tiles = 8 WMMA per K-step from 4 A-frags + 2 B-frags.
// ---------------------------------------------------------------------------
__global__ __launch_bounds__(256) void moe_expert_gemm(
    const float* __restrict__ x, const float* __restrict__ ew,
    const float* __restrict__ eb,
    const int* __restrict__ counts, const int* __restrict__ tok_list,
    const float* __restrict__ w_list, float* __restrict__ out)
{
    const int e    = blockIdx.x >> 7;            // / MAX_MT
    const int mt   = blockIdx.x & (MAX_MT - 1);
    const int cnt  = counts[e];
    const int row0 = mt * TM;
    if (row0 >= cnt) return;
    const int n0 = blockIdx.y * TN;

    __shared__ __bf16 As[TM][36];        // [m][k], padded, rows 4B-aligned
    __shared__ __bf16 Ws[TN][34];        // transposed: [n][k], padded
    __shared__ int    s_tok[TM];
    __shared__ float  s_w[TM];

    const int tid  = threadIdx.x;
    const int wv   = tid >> 5;
    const int lane = tid & 31;

    // Branch-free slot setup: invalid slots -> token 0 with weight 0.
    if (tid < TM) {
        const int r = row0 + tid;
        int   t = 0;
        float w = 0.f;
        if (r < cnt) {
            t = tok_list[e * MAXN + r];
            w = w_list [e * MAXN + r];
        }
        s_tok[tid] = t;
        s_w[tid]   = w;
    }
    __syncthreads();

    // ---- per-thread staging assignment, pointers hoisted out of K-loop ----
    // A: thread covers rows (tid>>4) + 16j (j=0..7), columns (tid&15)*2 .. +1
    const int am0 = tid >> 4;
    const int ak2 = (tid & 15) * 2;
    const float* ap[8];
#pragma unroll
    for (int j = 0; j < 8; ++j)
        ap[j] = x + (size_t)s_tok[am0 + 16 * j] * DIN + ak2;
    // W: thread covers k-row pairs {r0, r0+1} and {r0+16, r0+17}, cols wn..wn+3
    const int r0 = (tid >> 5) * 2;
    const int wn = (tid & 31) * 4;
    const float* wbase = ew + (size_t)e * DIN * DOUT + n0;
    const float* wp = wbase + (size_t)r0 * DOUT + wn;

    // wave tile coordinates: 2 (m) x 4 (n)
    const int wm  = wv >> 2;             // 0..1  -> 64-row half
    const int wn2 = wv & 3;              // 0..3  -> 32-col quarter
    const int arow = lane & 15;
    const int kA   = (lane < 16) ? 0 : 16;   // A-frag K half (matches B)
    const int bc0  = (wn2 << 5) + (lane & 15);
    const int bc1  = bc0 + 16;
    const int kB   = (lane < 16) ? 0 : 16;   // B-frag K half

    v8f c_frag[4][2] = {};

    float2 areg[8];
    float4 wreg[4];

    // prologue: load K-slab 0 into registers
#pragma unroll
    for (int j = 0; j < 8; ++j) areg[j] = *(const float2*)(ap[j]);
    wreg[0] = *(const float4*)(wp);
    wreg[1] = *(const float4*)(wp + DOUT);
    wreg[2] = *(const float4*)(wp + 16 * DOUT);
    wreg[3] = *(const float4*)(wp + 17 * DOUT);

    for (int kb = 0; kb < DIN / 32; ++kb) {
        const int k0 = kb * 32;

        // regs -> LDS: each pair one v_cvt_pk_bf16_f32 + one ds_store_b32
#pragma unroll
        for (int j = 0; j < 8; ++j)
            store_pair(&As[am0 + 16 * j][ak2], areg[j].x, areg[j].y);
        {
            const float* w0 = (const float*)&wreg[0];
            const float* w1 = (const float*)&wreg[1];
            const float* w2 = (const float*)&wreg[2];
            const float* w3 = (const float*)&wreg[3];
#pragma unroll
            for (int i = 0; i < 4; ++i) {
                store_pair(&Ws[wn + i][r0],      w0[i], w1[i]); // k = r0, r0+1
                store_pair(&Ws[wn + i][r0 + 16], w2[i], w3[i]); // k = +16, +17
            }
        }
        __syncthreads();

        // issue next K-slab's global loads; they complete under the WMMAs
        if (kb + 1 < DIN / 32) {
            const int kn = k0 + 32;
#pragma unroll
            for (int j = 0; j < 8; ++j)
                areg[j] = *(const float2*)(ap[j] + kn);
            const float* gp = wp + (size_t)kn * DOUT;
            wreg[0] = *(const float4*)(gp);
            wreg[1] = *(const float4*)(gp + DOUT);
            wreg[2] = *(const float4*)(gp + 16 * DOUT);
            wreg[3] = *(const float4*)(gp + 17 * DOUT);
            if (kn + 32 < DIN)
                __builtin_prefetch(gp + 32 * DOUT, 0, 1);  // slab after next
        }

        // fragments + 8 WMMA per wave (A reused over 2 N, B reused over 4 M)
        v16bf b0, b1;
#pragma unroll
        for (int j = 0; j < 16; ++j) {
            b0[j] = Ws[bc0][kB + j];
            b1[j] = Ws[bc1][kB + j];
        }
#pragma unroll
        for (int mi = 0; mi < 4; ++mi) {
            const int am = (wm << 6) + (mi << 4) + arow;
            v16bf a;
#pragma unroll
            for (int j = 0; j < 8; ++j) {
                a[j]     = As[am][kA + j];        // K = kA .. kA+7
                a[8 + j] = As[am][kA + 8 + j];    // K = kA+8 .. kA+15
            }
            c_frag[mi][0] = __builtin_amdgcn_wmma_f32_16x16x32_bf16(
                false, a, false, b0, (short)0, c_frag[mi][0], false, false);
            c_frag[mi][1] = __builtin_amdgcn_wmma_f32_16x16x32_bf16(
                false, a, false, b1, (short)0, c_frag[mi][1], false, false);
        }
        __syncthreads();
    }

    // Epilogue: bias, gate-weight scale, atomic add into output rows.
    // Exactly two fp32 adds land on each out element (commutative => deterministic).
    const int ncol0 = n0 + (wn2 << 5) + (lane & 15);
    const int ncol1 = ncol0 + 16;
    const float bias0 = eb[e * DOUT + ncol0];
    const float bias1 = eb[e * DOUT + ncol1];
    const int mbase = (lane < 16) ? 0 : 8;
#pragma unroll
    for (int mi = 0; mi < 4; ++mi) {
#pragma unroll
        for (int r = 0; r < 8; ++r) {
            const int m = (wm << 6) + (mi << 4) + mbase + r;
            const float w = s_w[m];
            if (w != 0.f) {
                float* orow = out + (size_t)s_tok[m] * DOUT;
                unsafeAtomicAdd(orow + ncol0, w * (c_frag[mi][0][r] + bias0));
                unsafeAtomicAdd(orow + ncol1, w * (c_frag[mi][1][r] + bias1));
            }
        }
    }
}

// ---------------------------------------------------------------------------
extern "C" void kernel_launch(void* const* d_in, const int* in_sizes, int n_in,
                              void* d_out, int out_size, void* d_ws, size_t ws_size,
                              hipStream_t stream)
{
    const float* x  = (const float*)d_in[0];   // [B, DIN]
    const float* gw = (const float*)d_in[1];   // [E, DIN]
    const float* gb = (const float*)d_in[2];   // [E]
    const float* ew = (const float*)d_in[3];   // [E, DIN, DOUT]
    const float* eb = (const float*)d_in[4];   // [E, DOUT]
    // d_in[5] = topk (==2, hardcoded)
    float* out = (float*)d_out;

    int*   counts   = (int*)d_ws;                                            // 256 B
    int*   tok_list = (int*)((char*)d_ws + 256);                             // 1 MB
    float* w_list   = (float*)((char*)d_ws + 256 + (size_t)NEXP * MAXN * 4); // 1 MB

    hipMemsetAsync(d_ws, 0, 256, stream);
    hipMemsetAsync(d_out, 0, (size_t)out_size * sizeof(float), stream);

    moe_gate_route<<<B_TOK / 8, 256, 0, stream>>>(x, gw, gb,
                                                  counts, tok_list, w_list);

    dim3 grid(NEXP * MAX_MT, DOUT / TN);
    moe_expert_gemm<<<grid, 256, 0, stream>>>(x, ew, eb,
                                              counts, tok_list, w_list, out);
}
